// GraphAttnEncoder_6150393168622
// MI455X (gfx1250) — matrix operand order repaired and checked by reading, not compiled, not run on hardware
//
#include <hip/hip_runtime.h>
#include <hip/hip_bf16.h>

typedef __attribute__((ext_vector_type(16))) _Float16 v16h;
typedef __attribute__((ext_vector_type(8)))  float    v8f;

#define N_NODES   50000
#define N_EDGES   800000
#define DIM       128     // HEADS*HID == IN == 128
#define HEADS     2
#define HID       64
#define NEG_SLOPE 0.2f
#define ROWS_PB   80      // rows per block = 5 subtiles of 16; 50000/80 = 625
#define SUBT      5

// ---------------- helpers: order-preserving float<->uint for atomic max ----
__device__ __forceinline__ unsigned f2ord(float f) {
    unsigned u = __float_as_uint(f);
    return (u & 0x80000000u) ? ~u : (u | 0x80000000u);
}
__device__ __forceinline__ float ord2f(unsigned u) {
    return (u & 0x80000000u) ? __uint_as_float(u & 0x7FFFFFFFu)
                             : __uint_as_float(~u);
}

// A-matrix (16-bit, 16x32) inverse layout: given kk (0..31) within a k-step,
// which lane-half and which fragment element holds it.
//   half = (kk >> 3) & 1 ;  elem = (kk & 7) | ((kk >= 16) ? 8 : 0)
__device__ __forceinline__ void a_frag_pos(int kk, int& half, int& elem) {
    half = (kk >> 3) & 1;
    elem = (kk & 7) + ((kk >= 16) ? 8 : 0);
}
// B-matrix (16-bit, 32x16) inverse layout: half = (kk >> 3) & 1 ; elem = (kk & 7) | ((kk >= 16) ? 8 : 0)
__device__ __forceinline__ void b_frag_pos(int kk, int& half, int& elem) {
    half = (kk >> 3) & 1;
    elem = (kk & 7) + ((kk >= 16) ? 8 : 0);
}

// ---------------- h = label_embed[labels] + features ------------------------
__global__ void embed_kernel(const float* __restrict__ features,
                             const int*   __restrict__ labels,
                             const float* __restrict__ label_embed,
                             float* __restrict__ h) {
    int gid = blockIdx.x * blockDim.x + threadIdx.x;
    if (gid >= N_NODES * DIM) return;
    int n = gid >> 7;            // /128
    int d = gid & 127;
    h[gid] = features[gid] + label_embed[labels[n] * DIM + d];
}

// ---------------- feat = h @ W  via v_wmma_f32_16x16x32_f16 -----------------
// block = 256 threads (8 waves). Block computes 80 rows x 128 cols.
// Wave w owns columns [16w, 16w+16) for all 5 M-subtiles; B fragments are
// loaded once into registers. LDS tiles are stored in *fragment order* so a
// lane's 16-f16 fragment is one contiguous 32B chunk (2x ds_load_b128).
__global__ __launch_bounds__(256) void gemm_wmma_kernel(
        const float* __restrict__ A,   // [N,128] row-major f32
        const float* __restrict__ W,   // [128,128] row-major f32
        float* __restrict__ C) {       // [N,128] f32
    // [subtile][kstep][lane][elem] and [kstep][wave][lane][elem]
    __shared__ __attribute__((aligned(32))) _Float16 AsFrag[SUBT][4][32][16]; // 20KB
    __shared__ __attribute__((aligned(32))) _Float16 BsFrag[4][8][32][16];    // 32KB

    const int tid = threadIdx.x;
    const int bm  = blockIdx.x;        // 80-row tile index

    // ---- stage whole W (128x128) into fragment-ordered LDS -------------
    // 4096 float4's, 16 per thread
    #pragma unroll
    for (int i = 0; i < 16; ++i) {
        int e4  = tid + 256 * i;       // 0..4095
        int k   = e4 >> 5;             // row of W (32 float4 per row)
        int c4  = e4 & 31;
        float4 f = ((const float4*)W)[k * 32 + c4];
        int t  = k >> 5, kk = k & 31, bh, be;
        b_frag_pos(kk, bh, be);
        float v[4] = {f.x, f.y, f.z, f.w};
        #pragma unroll
        for (int j = 0; j < 4; ++j) {
            int col  = c4 * 4 + j;
            int wv   = col >> 4;
            int lane = 16 * bh + (col & 15);
            BsFrag[t][wv][lane][be] = (_Float16)v[j];
        }
    }
    // ---- stage A tile (80x128) into fragment-ordered LDS ---------------
    // 2560 float4's, 10 per thread
    #pragma unroll
    for (int i = 0; i < 10; ++i) {
        int e4  = tid + 256 * i;       // 0..2559
        int row = e4 >> 5;             // 0..79
        int c4  = e4 & 31;
        float4 f = ((const float4*)A)[(bm * ROWS_PB + row) * 32 + c4];
        int sub = row >> 4, m = row & 15;
        float v[4] = {f.x, f.y, f.z, f.w};
        #pragma unroll
        for (int j = 0; j < 4; ++j) {
            int k = c4 * 4 + j;
            int t = k >> 5, kk = k & 31, ah, ae;
            a_frag_pos(kk, ah, ae);
            AsFrag[sub][t][16 * ah + m][ae] = (_Float16)v[j];
        }
    }
    __syncthreads();

    const int wave = tid >> 5;
    const int lane = tid & 31;
    const int m    = lane & 15;        // col within wave's 16-col tile
    const int half = lane >> 4;

    // B fragments for this wave: persistent in registers (4 x 8 VGPRs)
    v16h bfrag[4];
    #pragma unroll
    for (int t = 0; t < 4; ++t)
        bfrag[t] = *(const v16h*)&BsFrag[t][wave][lane][0];

    #pragma unroll
    for (int sub = 0; sub < SUBT; ++sub) {
        v8f acc = {};
        #pragma unroll
        for (int t = 0; t < 4; ++t) {  // K = 128 = 4 * 32
            v16h a = *(const v16h*)&AsFrag[sub][t][lane][0];
            acc = __builtin_amdgcn_wmma_f32_16x16x32_f16(
                      false, a, false, bfrag[t], (short)0, acc, false, false);
        }
        // C/D layout: VGPR r -> M = r + 8*half, N = lane&15
        int row0 = bm * ROWS_PB + sub * 16;
        #pragma unroll
        for (int r = 0; r < 8; ++r)
            C[(row0 + r + 8 * half) * DIM + 16 * wave + m] = acc[r];
    }
}

// ---------------- el/er per (node, head) ------------------------------------
__global__ void attn_dot_kernel(const float* __restrict__ feat,
                                const float* __restrict__ al,   // [H,HID]
                                const float* __restrict__ ar,   // [H,HID]
                                float* __restrict__ el,
                                float* __restrict__ er) {
    int gid = blockIdx.x * blockDim.x + threadIdx.x;
    if (gid >= N_NODES * HEADS) return;
    int n = gid / HEADS, h = gid % HEADS;
    const float* f = feat + n * DIM + h * HID;
    float sl = 0.f, sr = 0.f;
    #pragma unroll 4
    for (int d = 0; d < HID; ++d) {
        float x = f[d];
        sl += x * al[h * HID + d];
        sr += x * ar[h * HID + d];
    }
    el[gid] = sl;
    er[gid] = sr;
}

// ---------------- per-layer init (rst=0, m=-inf(ordered 0), s=0) ------------
__global__ void init_kernel(float* __restrict__ rst,
                            unsigned* __restrict__ m_u,
                            float* __restrict__ s_buf) {
    int gid = blockIdx.x * blockDim.x + threadIdx.x;
    if (gid < N_NODES * DIM) rst[gid] = 0.f;
    if (gid < N_NODES * HEADS) { m_u[gid] = 0u; s_buf[gid] = 0.f; }
}

// ---------------- edge pass 1: e = leakyrelu(el[src]+er[dst]); seg-max ------
__global__ void edge_max_kernel(const int* __restrict__ src,
                                const int* __restrict__ dst,
                                const float* __restrict__ el,
                                const float* __restrict__ er,
                                float* __restrict__ ebuf,
                                unsigned* __restrict__ m_u) {
    int gid = blockIdx.x * blockDim.x + threadIdx.x;
    if (gid >= N_EDGES * HEADS) return;
    int e = gid / HEADS, h = gid % HEADS;
    int s = src[e], d = dst[e];
    float x = el[s * HEADS + h] + er[d * HEADS + h];
    float le = (x > 0.f) ? x : NEG_SLOPE * x;
    ebuf[gid] = le;
    atomicMax(&m_u[d * HEADS + h], f2ord(le));
}

// ---------------- edge pass 2: ex = exp(e - m[dst]); seg-sum ----------------
__global__ void edge_exp_kernel(const int* __restrict__ dst,
                                const unsigned* __restrict__ m_u,
                                float* __restrict__ ebuf,   // in: e, out: ex
                                float* __restrict__ s_buf) {
    int gid = blockIdx.x * blockDim.x + threadIdx.x;
    if (gid >= N_EDGES * HEADS) return;
    int e = gid / HEADS, h = gid % HEADS;
    int d = dst[e];
    float m  = ord2f(m_u[d * HEADS + h]);
    float ex = __expf(ebuf[gid] - m);
    ebuf[gid] = ex;
    atomicAdd(&s_buf[d * HEADS + h], ex);
}

// ---------------- edge pass 3: rst[dst] += alpha * feat[src] ----------------
// 32 lanes per edge: lane c covers features [4c, 4c+4), head = c>>4.
__global__ void edge_scatter_kernel(const int* __restrict__ src,
                                    const int* __restrict__ dst,
                                    const float* __restrict__ exbuf,
                                    const float* __restrict__ s_buf,
                                    const float* __restrict__ feat,
                                    float* __restrict__ rst) {
    int gid = blockIdx.x * blockDim.x + threadIdx.x;
    if (gid >= N_EDGES * 32) return;
    int e = gid >> 5;
    int c = gid & 31;
    int h = c >> 4;
    int s = src[e], d = dst[e];
    float alpha = exbuf[e * HEADS + h] / s_buf[d * HEADS + h];
    float4 f = ((const float4*)feat)[s * 32 + c];
    float* out = rst + d * DIM + c * 4;
    atomicAdd(out + 0, alpha * f.x);
    atomicAdd(out + 1, alpha * f.y);
    atomicAdd(out + 2, alpha * f.z);
    atomicAdd(out + 3, alpha * f.w);
}

// ---------------- rst -> relu(rst + bias) -----------------------------------
__global__ void bias_relu_kernel(const float* __restrict__ rst,
                                 const float* __restrict__ bias,
                                 float* __restrict__ out) {
    int gid = blockIdx.x * blockDim.x + threadIdx.x;
    if (gid >= N_NODES * DIM) return;
    out[gid] = fmaxf(rst[gid] + bias[gid & 127], 0.f);
}

// ---------------------------------------------------------------------------
static void run_layer(const float* hin, const float* W, const float* al,
                      const float* ar, const float* bias, float* hout,
                      float* feat, float* rst, float* el, float* er,
                      unsigned* m_u, float* s_buf, float* ebuf,
                      const int* src, const int* dst, hipStream_t stream) {
    const int B = 256;
    gemm_wmma_kernel<<<N_NODES / ROWS_PB, B, 0, stream>>>(hin, W, feat);
    attn_dot_kernel<<<(N_NODES * HEADS + B - 1) / B, B, 0, stream>>>(feat, al, ar, el, er);
    init_kernel<<<(N_NODES * DIM + B - 1) / B, B, 0, stream>>>(rst, m_u, s_buf);
    edge_max_kernel<<<(N_EDGES * HEADS + B - 1) / B, B, 0, stream>>>(src, dst, el, er, ebuf, m_u);
    edge_exp_kernel<<<(N_EDGES * HEADS + B - 1) / B, B, 0, stream>>>(dst, m_u, ebuf, s_buf);
    edge_scatter_kernel<<<(N_EDGES * 32 + B - 1) / B, B, 0, stream>>>(src, dst, ebuf, s_buf, feat, rst);
    bias_relu_kernel<<<(N_NODES * DIM + B - 1) / B, B, 0, stream>>>(rst, bias, hout);
}

extern "C" void kernel_launch(void* const* d_in, const int* in_sizes, int n_in,
                              void* d_out, int out_size, void* d_ws, size_t ws_size,
                              hipStream_t stream) {
    const float* features    = (const float*)d_in[0];
    const int*   labels      = (const int*)  d_in[1];
    const int*   src         = (const int*)  d_in[2];
    const int*   dst         = (const int*)  d_in[3];
    const float* label_embed = (const float*)d_in[4];
    const float* W0  = (const float*)d_in[5];
    const float* al0 = (const float*)d_in[6];
    const float* ar0 = (const float*)d_in[7];
    const float* b0  = (const float*)d_in[8];
    const float* W1  = (const float*)d_in[9];
    const float* al1 = (const float*)d_in[10];
    const float* ar1 = (const float*)d_in[11];
    const float* b1  = (const float*)d_in[12];
    float* out = (float*)d_out;

    // workspace carve-up (all offsets keep 16B alignment)
    float*    h     = (float*)d_ws;                 // N*128
    float*    feat  = h    + N_NODES * DIM;         // N*128
    float*    rst   = feat + N_NODES * DIM;         // N*128
    float*    el    = rst  + N_NODES * DIM;         // N*H
    float*    er    = el   + N_NODES * HEADS;       // N*H
    unsigned* m_u   = (unsigned*)(er + N_NODES * HEADS); // N*H
    float*    s_buf = (float*)(m_u + N_NODES * HEADS);   // N*H
    float*    ebuf  = s_buf + N_NODES * HEADS;      // E*H

    const int B = 256;
    embed_kernel<<<(N_NODES * DIM + B - 1) / B, B, 0, stream>>>(features, labels, label_embed, h);

    // layer 0: h -> h (in place after GEMM is done with its input)
    run_layer(h, W0, al0, ar0, b0, h, feat, rst, el, er, m_u, s_buf, ebuf, src, dst, stream);
    // layer 1: h -> d_out
    run_layer(h, W1, al1, ar1, b1, out, feat, rst, el, er, m_u, s_buf, ebuf, src, dst, stream);
}
